// Gate_17712445128840
// MI455X (gfx1250) — matrix operand order repaired and checked by reading, not compiled, not run on hardware
//
#include <hip/hip_runtime.h>

// ---------------------------------------------------------------------------
// MoE router gate for MI455X (gfx1250, wave32).
//   Pass 1: scores = sigmoid(x @ W^T) via v_wmma_f32_16x16x32_bf16 -> d_ws
//   Pass 2: grouped top-k routing (one wave32 per token) -> d_out
// ---------------------------------------------------------------------------

typedef __attribute__((ext_vector_type(16))) __bf16 v16bf;
typedef __attribute__((ext_vector_type(8)))  __bf16 v8bf;
typedef __attribute__((ext_vector_type(8)))  float  v8f;

constexpr int T_ = 16384;   // tokens
constexpr int D_ = 7168;    // d_model
constexpr int E_ = 256;     // experts
constexpr int K_ = 8;       // activated experts

constexpr int BM = 128;     // tile M
constexpr int BN = 128;     // tile N
constexpr int BK = 64;      // tile K (two wmma k-steps of 32)
constexpr int SA = BK + 8;  // padded LDS row stride (bf16 units) -> conflict-free

__device__ __forceinline__ unsigned short f2bf(float f) {
  // round-to-nearest-even f32 -> bf16
  unsigned u = __float_as_uint(f);
  u += 0x7FFFu + ((u >> 16) & 1u);
  return (unsigned short)(u >> 16);
}

__global__ __launch_bounds__(256)
void gemm_sigmoid_kernel(const float* __restrict__ X,   // [T, D]
                         const float* __restrict__ W,   // [E, D]  (K-contiguous = B^T layout)
                         float* __restrict__ S) {       // [T, E]  sigmoid scores
  __shared__ __align__(16) unsigned short As[BM * SA];
  __shared__ __align__(16) unsigned short Bs[BN * SA];

  const int tid  = threadIdx.x;
  const int lane = tid & 31;
  const int wave = tid >> 5;        // 8 waves
  const int wm   = wave >> 2;       // 0..1 -> 64 rows each
  const int wn   = wave & 3;        // 0..3 -> 32 cols each
  const int m0   = blockIdx.y * BM;
  const int n0   = blockIdx.x * BN;
  const int lhalf = lane >> 4;      // 0 | 1
  const int l16   = lane & 15;

  v8f acc[4][2] = {};

  for (int kb = 0; kb < D_; kb += BK) {
    // ---- global f32 -> LDS bf16 tile loads: 128x64 each, float4 vectorized
    #pragma unroll
    for (int it = 0; it < 8; ++it) {
      const int idx = tid + it * 256;
      const int r   = idx >> 4;     // 0..127
      const int k4  = idx & 15;     // 0..15 (float4 within BK)
      const float4 va = *(reinterpret_cast<const float4*>(X + (size_t)(m0 + r) * D_ + kb) + k4);
      ushort4 pa;
      pa.x = f2bf(va.x); pa.y = f2bf(va.y); pa.z = f2bf(va.z); pa.w = f2bf(va.w);
      *reinterpret_cast<ushort4*>(&As[r * SA + k4 * 4]) = pa;
      const float4 vb = *(reinterpret_cast<const float4*>(W + (size_t)(n0 + r) * D_ + kb) + k4);
      ushort4 pb;
      pb.x = f2bf(vb.x); pb.y = f2bf(vb.y); pb.z = f2bf(vb.z); pb.w = f2bf(vb.w);
      *reinterpret_cast<ushort4*>(&Bs[r * SA + k4 * 4]) = pb;
    }
    __syncthreads();

    #pragma unroll
    for (int kk = 0; kk < BK; kk += 32) {
      // A frag (16x32 bf16): lanes 0-15 -> K 0-7 & 16-23; lanes 16-31 -> K 8-15 & 24-31
      const int aks = kk + (lhalf << 3);
      // B frag (32x16 bf16): lanes 0-15 -> K 0-7 & 16-23; lanes 16-31 -> K 8-15 & 24-31
      const int bks = kk + (lhalf << 3);

      v16bf afrag[4];
      #pragma unroll
      for (int i = 0; i < 4; ++i) {
        const int row = wm * 64 + i * 16 + l16;
        v8bf lo = __builtin_bit_cast(v8bf, *reinterpret_cast<const uint4*>(&As[row * SA + aks]));
        v8bf hi = __builtin_bit_cast(v8bf, *reinterpret_cast<const uint4*>(&As[row * SA + aks + 16]));
        afrag[i] = __builtin_shufflevector(lo, hi, 0,1,2,3,4,5,6,7,8,9,10,11,12,13,14,15);
      }
      v16bf bfrag[2];
      #pragma unroll
      for (int j = 0; j < 2; ++j) {
        const int col = wn * 32 + j * 16 + l16;
        v8bf lo = __builtin_bit_cast(v8bf, *reinterpret_cast<const uint4*>(&Bs[col * SA + bks]));
        v8bf hi = __builtin_bit_cast(v8bf, *reinterpret_cast<const uint4*>(&Bs[col * SA + bks + 16]));
        bfrag[j] = __builtin_shufflevector(lo, hi, 0,1,2,3,4,5,6,7,8,9,10,11,12,13,14,15);
      }

      #pragma unroll
      for (int i = 0; i < 4; ++i)
        #pragma unroll
        for (int j = 0; j < 2; ++j)
          acc[i][j] = __builtin_amdgcn_wmma_f32_16x16x32_bf16(
              false, afrag[i], false, bfrag[j], (short)0, acc[i][j], false, false);
    }
    __syncthreads();
  }

  // ---- epilogue: sigmoid, store f32 scores
  // C/D layout: VGPR v -> M = v (lanes 0-15) or v+8 (lanes 16-31); N = lane&15
  #pragma unroll
  for (int i = 0; i < 4; ++i)
    #pragma unroll
    for (int j = 0; j < 2; ++j)
      #pragma unroll
      for (int v = 0; v < 8; ++v) {
        const int row = m0 + wm * 64 + i * 16 + v + (lhalf << 3);
        const int col = n0 + wn * 32 + j * 16 + l16;
        const float z = acc[i][j][v];
        S[(size_t)row * E_ + col] = 1.0f / (1.0f + __expf(-z));
      }
}

// ---------------------------------------------------------------------------
// One wave32 per token: grouped top-2 / top-4-groups / global top-8.
// Lane l owns experts e = 32*j + l (j = group index 0..7).
// ---------------------------------------------------------------------------
__global__ __launch_bounds__(256)
void router_topk_kernel(const float* __restrict__ S,     // [T, E]
                        const float* __restrict__ bias,  // [E]
                        float* __restrict__ w_out,       // [T, K]
                        int* __restrict__ i_out) {       // [T, K]
  const int lane = threadIdx.x & 31;
  const int wave = threadIdx.x >> 5;
  const int t    = blockIdx.x * 8 + wave;
  constexpr float NEG = -1.0e30f;

  float orig[8], sb[8];
  #pragma unroll
  for (int j = 0; j < 8; ++j) {
    const int e = 32 * j + lane;
    const float s = S[(size_t)t * E_ + e];
    orig[j] = s;
    sb[j]   = s + bias[e];
  }

  // group score = sum of top-2 within the group's 32 lanes
  float gs[8];
  #pragma unroll
  for (int j = 0; j < 8; ++j) {
    float hi = sb[j], lo = NEG;
    #pragma unroll
    for (int m = 1; m < 32; m <<= 1) {
      const float ohi = __shfl_xor(hi, m, 32);
      const float olo = __shfl_xor(lo, m, 32);
      const float nhi = fmaxf(hi, ohi);
      const float nlo = fmaxf(fminf(hi, ohi), fmaxf(lo, olo));
      hi = nhi; lo = nlo;
    }
    gs[j] = hi + lo;
  }

  // top-4 groups (strict > keeps lowest index on ties, matching lax.top_k)
  unsigned kmask = 0;
  #pragma unroll
  for (int s = 0; s < 4; ++s) {
    float best = -2.0e30f; int bg = 0;
    #pragma unroll
    for (int g = 0; g < 8; ++g)
      if (!((kmask >> g) & 1u) && gs[g] > best) { best = gs[g]; bg = g; }
    kmask |= 1u << bg;
  }
  #pragma unroll
  for (int j = 0; j < 8; ++j)
    if (!((kmask >> j) & 1u)) sb[j] = NEG;

  // global top-8 over masked biased scores; gather ORIGINAL sigmoid scores
  float wk[8]; int ik[8]; float wsum = 0.0f;
  #pragma unroll
  for (int k = 0; k < 8; ++k) {
    float bv = -2.0e30f; int be = 1 << 30; float bo = 0.0f;
    #pragma unroll
    for (int j = 0; j < 8; ++j)
      if (sb[j] > bv) { bv = sb[j]; be = 32 * j + lane; bo = orig[j]; }
    #pragma unroll
    for (int m = 1; m < 32; m <<= 1) {
      const float ov = __shfl_xor(bv, m, 32);
      const int   oe = __shfl_xor(be, m, 32);
      const float oo = __shfl_xor(bo, m, 32);
      if (ov > bv || (ov == bv && oe < be)) { bv = ov; be = oe; bo = oo; }
    }
    wk[k] = bo; ik[k] = be; wsum += bo;
    if ((be & 31) == lane) sb[be >> 5] = NEG;   // owner retires winner
  }

  if (lane == 0) {
    const float scale = 2.5f / wsum;
    #pragma unroll
    for (int k = 0; k < 8; ++k) {
      w_out[(size_t)t * K_ + k] = wk[k] * scale;
      i_out[(size_t)t * K_ + k] = ik[k];
    }
  }
}

extern "C" void kernel_launch(void* const* d_in, const int* in_sizes, int n_in,
                              void* d_out, int out_size, void* d_ws, size_t ws_size,
                              hipStream_t stream) {
  (void)in_sizes; (void)n_in; (void)out_size; (void)ws_size;
  const float* x    = (const float*)d_in[0];   // [T, D]
  const float* w    = (const float*)d_in[1];   // [E, D]
  const float* bias = (const float*)d_in[2];   // [E]

  float* scores = (float*)d_ws;                          // [T, E] f32 = 16 MB
  float* w_out  = (float*)d_out;                         // [T, K] f32
  int*   i_out  = (int*)((float*)d_out + (size_t)T_ * K_); // [T, K] i32

  dim3 grid_gemm(E_ / BN, T_ / BM);   // (2, 128)
  gemm_sigmoid_kernel<<<grid_gemm, 256, 0, stream>>>(x, w, scores);
  router_topk_kernel<<<T_ / 8, 256, 0, stream>>>(scores, bias, w_out, i_out);
}
